// ScaledDotProduct_50465865728114
// MI455X (gfx1250) — matrix operand order repaired and checked by reading, not compiled, not run on hardware
//
#include <hip/hip_runtime.h>

typedef _Float16 v16h __attribute__((ext_vector_type(16)));
typedef _Float16 v8h  __attribute__((ext_vector_type(8)));
typedef _Float16 v4h  __attribute__((ext_vector_type(4)));
typedef float    v8f  __attribute__((ext_vector_type(8)));

namespace {

constexpr int kB = 4, kH = 16, kS = 2048, kD = 64;
constexpr int kBK  = 64;          // keys per streamed block
constexpr int kQB  = 128;         // queries per workgroup (8 waves x 16)
constexpr int kLDP = 72;          // padded LDS row pitch (halves)
constexpr int kNKB = kS / kBK;    // 32 key blocks
// 1/sqrt(64) * log2(e), folded into K at f16-conversion time (log2 domain)
constexpr float kScale = 0.125f * 1.4426950408889634f;
constexpr size_t kElems  = (size_t)kB * kH * kS * kD;          // 8388608
constexpr size_t kWsNeed = 2 * kElems * sizeof(_Float16);      // K(f16)+Vt(f16)

union H16 { v16h v; v8h h[2]; };

// A-layout (16-bit A 16x32): lanes 0-15 hold K-chunks {0..7},{16..23};
// lanes 16-31 hold {8..15},{24..31}.  dhi = 8*(lane>>4).
__device__ inline v16h lds_load_a(const _Float16* p, int dhi) {
  H16 u;
  u.h[0] = *(const v8h*)(p + dhi);
  u.h[1] = *(const v8h*)(p + dhi + 16);
  return u.v;
}

// B-layout (16-bit B 32x16): lane holds column (lane&15); lanes 0-15 hold
// K-chunks {0..7},{16..23}; lanes 16-31 hold {8..15},{24..31}.  koff = 16*(lane>>4).
__device__ inline v16h lds_load_b(const _Float16* p, int koff) {
  H16 u;
  u.h[0] = *(const v8h*)(p + koff / 2);
  u.h[1] = *(const v8h*)(p + koff / 2 + 16);
  return u.v;
}

// Flat LDS pointer -> 32-bit group-segment offset.
__device__ inline unsigned lds_addr32(const void* p) {
  return (unsigned)(unsigned long long)(uintptr_t)p;
}

// Async DMA: 16B per lane, global (saddr + 32-bit voffset) -> LDS.
// Tracked by ASYNCcnt; no VGPR data round-trip.
__device__ inline void async_cp16(unsigned lds, const void* sbase, unsigned voff) {
  asm volatile("global_load_async_to_lds_b128 %0, %1, %2 offset:0"
               :: "v"(lds), "v"(voff), "s"(sbase) : "memory");
}

// Raw v_exp_f32: argument is always <= 0 here, and underflow-to-zero is the
// mathematically desired softmax behavior -- skip OCML's subnormal guard.
__device__ inline float fast_exp2(float x) {
#if __has_builtin(__builtin_amdgcn_exp2f)
  return __builtin_amdgcn_exp2f(x);
#else
  return exp2f(x);
#endif
}

// DPP16 row-rotate (16-lane row == one half-wave).  Pure VALU, no LDS.
template <int CTRL>
__device__ inline float dpp_rot(float x) {
  return __int_as_float(__builtin_amdgcn_update_dpp(
      0, __float_as_int(x), CTRL, 0xf, 0xf, true));
}
// Full 16-lane max, result replicated to all lanes of the row.
__device__ inline float row_max16(float x) {
  x = fmaxf(x, dpp_rot<0x128>(x));   // row_ror:8
  x = fmaxf(x, dpp_rot<0x124>(x));   // row_ror:4
  x = fmaxf(x, dpp_rot<0x122>(x));   // row_ror:2
  x = fmaxf(x, dpp_rot<0x121>(x));   // row_ror:1
  return x;
}

// ---- prep: K -> f16 (pre-scaled, row-major), V -> f16 transposed [bh][d][s] ----
__global__ __launch_bounds__(256)
void prep_f16(const float* __restrict__ Kg, const float* __restrict__ Vg,
              _Float16* __restrict__ Kh, _Float16* __restrict__ Vth)
{
  const size_t flat = ((size_t)blockIdx.x * 256 + threadIdx.x) * 4;
  const int bh  = (int)(flat >> 17);          // / (S*D)
  const int rem = (int)(flat & (size_t)(kS * kD - 1));
  const int key = rem >> 6;
  const int dp  = rem & 63;
  float4 kf = *(const float4*)(Kg + flat);
  float4 vf = *(const float4*)(Vg + flat);
  v4h hk = { (_Float16)(kf.x * kScale), (_Float16)(kf.y * kScale),
             (_Float16)(kf.z * kScale), (_Float16)(kf.w * kScale) };
  *(v4h*)(Kh + flat) = hk;
  _Float16* vb = Vth + (size_t)bh * kS * kD;
  vb[(dp + 0) * kS + key] = (_Float16)vf.x;
  vb[(dp + 1) * kS + key] = (_Float16)vf.y;
  vb[(dp + 2) * kS + key] = (_Float16)vf.z;
  vb[(dp + 3) * kS + key] = (_Float16)vf.w;
}

// ---- flash-attention forward; ASYNC=true uses f16 ws + async double-buffer ----
template <bool ASYNC>
__global__ __launch_bounds__(256)
void fa_fwd(const float* __restrict__ Qg, const float* __restrict__ Kg,
            const float* __restrict__ Vg, const _Float16* __restrict__ Kh,
            const _Float16* __restrict__ Vth, float* __restrict__ Og)
{
  __shared__ _Float16 Kbuf[2][kBK * kLDP];   // [key][d] (pre-scaled f16)
  __shared__ _Float16 Vbuf[2][kD * kLDP];    // [d][key] (transposed f16)
  __shared__ _Float16 Pl[8][16 * kLDP];      // per-wave P tile [row][key]

  const int tid  = threadIdx.x;
  const int lane = tid & 31;
  const int wave = tid >> 5;
  const int half = lane >> 4;
  const int lx   = lane & 15;
  const int dhi  = half * 8;    // A-layout chunk offset
  const int koff = half * 16;   // B-layout K offset

  const long base = (long)blockIdx.y * kS * kD;
  const float* qp = Qg + base;
  float* op = Og + base;
  const int q0 = blockIdx.x * kQB;

  const _Float16* khb = nullptr;
  const _Float16* vtb = nullptr;
  const float* kp = nullptr;
  const float* vp = nullptr;
  if constexpr (ASYNC) { khb = Kh + base; vtb = Vth + base; }
  else                 { kp = Kg + base;  vp = Vg + base; }

  // Issue one 64-key block: 4 async b128 instructions per wave.
  auto issue_block = [&](int kb, int buf) {
    const int k0 = kb * kBK;
    #pragma unroll
    for (int i = 0; i < 2; ++i) {
      const int id  = tid + i * 256;   // 0..511
      const int row = id >> 3;         // key (K) / d (V)
      const int c   = id & 7;          // 16B chunk within row
      async_cp16(lds_addr32(&Kbuf[buf][row * kLDP + c * 8]), khb,
                 (unsigned)(((k0 + row) * kD + c * 8) * 2));
      async_cp16(lds_addr32(&Vbuf[buf][row * kLDP + c * 8]), vtb,
                 (unsigned)((row * kS + k0 + c * 8) * 2));
    }
  };

  // ---- load this wave's 16-query Q tile straight into WMMA A-layout ----
  v16h aq[2];
  {
    const long qrow = q0 + wave * 16 + lx;
    #pragma unroll
    for (int n = 0; n < 2; ++n) {
      H16 u;
      #pragma unroll
      for (int c = 0; c < 2; ++c) {
        const float* s = qp + qrow * kD + n * 32 + dhi + c * 16;
        float4 f0 = *(const float4*)(s);
        float4 f1 = *(const float4*)(s + 4);
        u.v[c*8+0] = (_Float16)f0.x; u.v[c*8+1] = (_Float16)f0.y;
        u.v[c*8+2] = (_Float16)f0.z; u.v[c*8+3] = (_Float16)f0.w;
        u.v[c*8+4] = (_Float16)f1.x; u.v[c*8+5] = (_Float16)f1.y;
        u.v[c*8+6] = (_Float16)f1.z; u.v[c*8+7] = (_Float16)f1.w;
      }
      aq[n] = u.v;
    }
  }

  // All-ones B matrix: rowsum(P) = P @ 1 via the matrix pipe.
  v16h ones;
  #pragma unroll
  for (int i = 0; i < 16; ++i) ones[i] = (_Float16)1.0f;

  float m[8], l[8];
  v8f acc[4];
  #pragma unroll
  for (int r = 0; r < 8; ++r) { m[r] = -3.0e38f; l[r] = 0.0f; }
  #pragma unroll
  for (int j = 0; j < 4; ++j) acc[j] = (v8f){0,0,0,0,0,0,0,0};

  if constexpr (ASYNC) issue_block(0, 0);

  for (int kb = 0; kb < kNKB; ++kb) {
    const _Float16* Kt;
    const _Float16* Vt;
    if constexpr (ASYNC) {
      if (kb + 1 < kNKB) {
        issue_block(kb + 1, (kb + 1) & 1);
        // async loads complete in order: <=4 outstanding == block kb landed
        asm volatile("s_wait_asynccnt 0x4" ::: "memory");
      } else {
        asm volatile("s_wait_asynccnt 0x0" ::: "memory");
      }
      Kt = &Kbuf[kb & 1][0];
      Vt = &Vbuf[kb & 1][0];
    } else {
      // fallback: synchronous f32 stage-in with conversion (scale folded)
      const int k0 = kb * kBK;
      #pragma unroll
      for (int i = 0; i < 4; ++i) {
        const int f4i = tid + i * 256;
        const int key = f4i >> 4;
        const int dp  = (f4i & 15) * 4;
        float4 kf = *(const float4*)(kp + (long)(k0 + key) * kD + dp);
        float4 vf = *(const float4*)(vp + (long)(k0 + key) * kD + dp);
        _Float16* kd = &Kbuf[0][key * kLDP + dp];
        kd[0] = (_Float16)(kf.x * kScale); kd[1] = (_Float16)(kf.y * kScale);
        kd[2] = (_Float16)(kf.z * kScale); kd[3] = (_Float16)(kf.w * kScale);
        Vbuf[0][(dp+0) * kLDP + key] = (_Float16)vf.x;
        Vbuf[0][(dp+1) * kLDP + key] = (_Float16)vf.y;
        Vbuf[0][(dp+2) * kLDP + key] = (_Float16)vf.z;
        Vbuf[0][(dp+3) * kLDP + key] = (_Float16)vf.w;
        if (kb + 1 < kNKB) {
          __builtin_prefetch(kp + (long)(k0 + kBK + key) * kD + dp, 0, 3);
          __builtin_prefetch(vp + (long)(k0 + kBK + key) * kD + dp, 0, 3);
        }
      }
      Kt = &Kbuf[0][0];
      Vt = &Vbuf[0][0];
    }
    __syncthreads();

    // ---- S = Q K^T over 4 sub-tiles of 16 keys (already log2-scaled) ----
    v8f sc[4];
    #pragma unroll
    for (int t = 0; t < 4; ++t) {
      const _Float16* kr = Kt + (t * 16 + lx) * kLDP;
      v8f s = (v8f){0,0,0,0,0,0,0,0};
      s = __builtin_amdgcn_wmma_f32_16x16x32_f16(false, aq[0], false,
            lds_load_b(kr, koff), (short)0, s, false, false);
      s = __builtin_amdgcn_wmma_f32_16x16x32_f16(false, aq[1], false,
            lds_load_b(kr + 32, koff), (short)0, s, false, false);
      sc[t] = s;
    }

    // ---- online softmax: DPP16 row-max (pure VALU), raw v_exp_f32 ----
    float al8[8];
    #pragma unroll
    for (int r = 0; r < 8; ++r) {
      float mx = fmaxf(fmaxf(sc[0][r], sc[1][r]), fmaxf(sc[2][r], sc[3][r]));
      mx = row_max16(mx);
      const float mn = fmaxf(m[r], mx);
      al8[r] = fast_exp2(m[r] - mn);
      m[r] = mn;
      #pragma unroll
      for (int t = 0; t < 4; ++t) sc[t][r] = fast_exp2(sc[t][r] - mn);
      #pragma unroll
      for (int j = 0; j < 4; ++j) acc[j][r] *= al8[r];   // rescale running O
    }

    // ---- C-layout -> A-layout for P via per-wave LDS tile ----
    _Float16* pw = &Pl[wave][0];
    #pragma unroll
    for (int t = 0; t < 4; ++t)
      #pragma unroll
      for (int r = 0; r < 8; ++r)
        pw[(half * 8 + r) * kLDP + t * 16 + lx] = (_Float16)sc[t][r];
    // per-wave DS ops are in-order; just drain DScnt before re-reading
    asm volatile("s_wait_dscnt 0x0" ::: "memory");

    // ---- O += P V, and row-sums l += P @ 1 on the matrix pipe ----
    v8f rsum = (v8f){0,0,0,0,0,0,0,0};
    #pragma unroll
    for (int kt = 0; kt < 2; ++kt) {
      const v16h ap = lds_load_a(&pw[lx * kLDP + kt * 32], dhi);
      rsum = __builtin_amdgcn_wmma_f32_16x16x32_f16(false, ap, false, ones,
                                                    (short)0, rsum, false, false);
      #pragma unroll
      for (int j = 0; j < 4; ++j) {
        const v16h bv = lds_load_b(&Vt[(j * 16 + lx) * kLDP + kt * 32], koff);
        acc[j] = __builtin_amdgcn_wmma_f32_16x16x32_f16(false, ap, false, bv,
                                                        (short)0, acc[j],
                                                        false, false);
      }
    }
    #pragma unroll
    for (int r = 0; r < 8; ++r) l[r] = l[r] * al8[r] + rsum[r];

    __syncthreads();   // protect LDS tiles before next block lands/stages
  }

  // ---- normalize and write O (C-layout: lane half picks row group) ----
  #pragma unroll
  for (int r = 0; r < 8; ++r) {
    const float inv = 1.0f / l[r];
    const long row = q0 + wave * 16 + half * 8 + r;
    #pragma unroll
    for (int j = 0; j < 4; ++j)
      op[row * kD + j * 16 + lx] = acc[j][r] * inv;
  }
}

} // namespace

extern "C" void kernel_launch(void* const* d_in, const int* in_sizes, int n_in,
                              void* d_out, int out_size, void* d_ws, size_t ws_size,
                              hipStream_t stream) {
  const float* q = (const float*)d_in[0];
  const float* k = (const float*)d_in[1];
  const float* v = (const float*)d_in[2];
  float* o = (float*)d_out;
  dim3 grid(kS / kQB, kB * kH);   // (16, 64) workgroups
  dim3 block(256);                // 8 waves (wave32)
  if (ws_size >= kWsNeed) {
    _Float16* kh = (_Float16*)d_ws;
    _Float16* vt = kh + kElems;
    hipLaunchKernelGGL(prep_f16, dim3((unsigned)(kElems / 4 / 256)), dim3(256),
                       0, stream, k, v, kh, vt);
    hipLaunchKernelGGL(fa_fwd<true>, grid, block, 0, stream,
                       q, k, v, kh, vt, o);
  } else {
    hipLaunchKernelGGL(fa_fwd<false>, grid, block, 0, stream,
                       q, k, v, (const _Float16*)nullptr,
                       (const _Float16*)nullptr, o);
  }
}
